// LFA_72464688218282
// MI455X (gfx1250) — matrix operand order repaired and checked by reading, not compiled, not run on hardware
//
#include <hip/hip_runtime.h>
#include <hip/hip_bf16.h>

// LFA (Point-Transformer local feature aggregation), MI455X / gfx1250, wave32.
// Kernel 1: x = f@W1^T+b1 ; q,k,v = x@W{q,k,v}^T + b  (WMMA f16, 16 pts/wave)
// Kernel 2: fully fused per-point attention, one wave32 per point:
//   - async gather of 16 k-rows + 16 v-rows (GLOBAL_LOAD_ASYNC_TO_LDS_B128,
//     ASYNCcnt) overlapped with the pos-MLP
//   - 2x LN+ReLU, two (16x64)@(64x64) v_wmma_f32_16x16x32_f16 GEMMs
//   - column softmax over K, weighted sum, BN-ReLU epilogue.

#define BQ   2
#define NPT  20480
#define KNN  16
#define DIN  32
#define CCH  64
#define EPS  1e-5f
#define WPB  2          // waves per block in kernel 2

typedef __attribute__((ext_vector_type(16))) _Float16 v16h;
typedef __attribute__((ext_vector_type(8)))  _Float16 v8h;
typedef __attribute__((ext_vector_type(8)))  float    v8f;

__device__ __forceinline__ v16h frag_cat(v8h a, v8h b) {
    return __builtin_shufflevector(a, b, 0,1,2,3,4,5,6,7,8,9,10,11,12,13,14,15);
}

// A-fragment (16x32 f16, M x K) from row-major f16 LDS tile.
// Contiguous k order: lanes 0-15 m=0..15 hold K {0-15}; lanes 16-31 hold K {16-31}.
__device__ __forceinline__ v16h load_a_frag(const _Float16* t, int strideH, int kbase) {
    int lane = threadIdx.x & 31;
    int m    = lane & 15;
    int hi   = lane >> 4;
    const _Float16* row = t + m * strideH + kbase;
    v8h a = *(const v8h*)(row + (hi ? 16 : 0));
    v8h b = *(const v8h*)(row + (hi ? 24 : 8));
    return frag_cat(a, b);
}

// B-fragment from pre-swizzled LDS (frag stored contiguous per lane: [lane][16]).
__device__ __forceinline__ v16h load_b_frag(const _Float16* w) {
    int lane = threadIdx.x & 31;
    const _Float16* p = w + lane * 16;
    v8h a = *(const v8h*)(p);
    v8h b = *(const v8h*)(p + 8);
    return frag_cat(a, b);
}

__device__ __forceinline__ v8f wmma_f16(v16h a, v16h b, v8f c) {
    return __builtin_amdgcn_wmma_f32_16x16x32_f16(false, a, false, b, (short)0, c,
                                                  false, false);
}

// Generic "shared" pointer: low 32 bits are the LDS byte address (ISA 10.2).
__device__ __forceinline__ unsigned lds_off(const void* p) {
    return (unsigned)(unsigned long long)p;
}

// Async gather of one 16-byte chunk: global -> LDS, tracked by ASYNCcnt.
__device__ __forceinline__ void async_copy_b128(unsigned lds_dst,
                                                unsigned long long gaddr) {
    asm volatile("global_load_async_to_lds_b128 %0, %1, off"
                 :: "v"(lds_dst), "v"(gaddr) : "memory");
}

__device__ __forceinline__ void wait_asynccnt0() {
    asm volatile("s_wait_asynccnt 0x0" ::: "memory");
}

// ------------------------------------------------------------------ kernel 1
// 4 waves / block, each wave handles 16 consecutive points.
template <typename OUT>
__device__ __forceinline__ void gemm_qkv16(const _Float16* xh, const _Float16* wf,
                                           const float* bias, OUT* dst) {
    int lane = threadIdx.x & 31, nn = lane & 15, hi = lane >> 4;
    v16h a0 = load_a_frag(xh, 64, 0);
    v16h a1 = load_a_frag(xh, 64, 32);
#pragma unroll
    for (int nt = 0; nt < 4; ++nt) {
        float bb = bias[nt * 16 + nn];
        v8f acc;
#pragma unroll
        for (int r = 0; r < 8; ++r) acc[r] = bb;
        acc = wmma_f16(a0, load_b_frag(wf + (nt * 2 + 0) * 512), acc);
        acc = wmma_f16(a1, load_b_frag(wf + (nt * 2 + 1) * 512), acc);
#pragma unroll
        for (int r = 0; r < 8; ++r)
            dst[(r + hi * 8) * CCH + nt * 16 + nn] = (OUT)acc[r];
    }
}

__global__ __launch_bounds__(128) void lfa_qkv_kernel(
    const float* __restrict__ feat, const float* __restrict__ W1,
    const float* __restrict__ b1,  const float* __restrict__ Wq,
    const float* __restrict__ bq,  const float* __restrict__ Wk,
    const float* __restrict__ bk,  const float* __restrict__ Wv,
    const float* __restrict__ bv,
    float* __restrict__ qb, _Float16* __restrict__ kb, _Float16* __restrict__ vb) {
    // frags: W1 -> 0..3 (nt, K=32 single chunk); Wq 4..11, Wk 12..19, Wv 20..27
    __shared__ __align__(16) _Float16 wfrag[28 * 512];
    __shared__ __align__(16) _Float16 fA[4][16 * 32];
    __shared__ __align__(16) _Float16 xh[4][16 * 64];

    int tid = threadIdx.x;
    for (int idx = tid; idx < 28 * 512; idx += 128) {
        int f = idx >> 9, r = idx & 511, lane = r >> 4, j = r & 15;
        int nn = lane & 15, kk = ((lane >> 4) ? 16 : 0) + j;
        float v;
        if (f < 4) {
            v = W1[(f * 16 + nn) * DIN + kk];                 // K = 32, kc = 0
        } else {
            int g = f - 4, wi = g >> 3, h = g & 7, nt = h >> 1, kc = h & 1;
            const float* W = (wi == 0) ? Wq : ((wi == 1) ? Wk : Wv);
            v = W[(nt * 16 + nn) * CCH + kc * 32 + kk];
        }
        wfrag[idx] = (_Float16)v;
    }
    __syncthreads();

    int warp = tid >> 5, lane = tid & 31;
    int bn0  = (blockIdx.x * 4 + warp) * 16;          // first flat point index
    int b    = bn0 / NPT, n0 = bn0 % NPT;
    const float* fbase = feat + (size_t)b * DIN * NPT;
    _Float16* myA = fA[warp];
    _Float16* myX = xh[warp];

    // stage f (16 x 32) as f16 (transpose from (DIN, N) layout)
    for (int i = lane; i < 16 * 32; i += 32) {
        int m = i >> 5, d = i & 31;
        myA[m * 32 + d] = (_Float16)fbase[(size_t)d * NPT + n0 + m];
    }

    // x = f @ W1^T + b1  -> f16 tile
    {
        int nn = lane & 15, hi = lane >> 4;
        v16h af = load_a_frag(myA, 32, 0);
#pragma unroll
        for (int nt = 0; nt < 4; ++nt) {
            float bb = b1[nt * 16 + nn];
            v8f acc;
#pragma unroll
            for (int r = 0; r < 8; ++r) acc[r] = bb;
            acc = wmma_f16(af, load_b_frag(wfrag + nt * 512), acc);
#pragma unroll
            for (int r = 0; r < 8; ++r)
                myX[(r + hi * 8) * 64 + nt * 16 + nn] = (_Float16)acc[r];
        }
    }

    gemm_qkv16<float>   (myX, wfrag + 4  * 512, bq, qb + (size_t)bn0 * CCH);
    gemm_qkv16<_Float16>(myX, wfrag + 12 * 512, bk, kb + (size_t)bn0 * CCH);
    gemm_qkv16<_Float16>(myX, wfrag + 20 * 512, bv, vb + (size_t)bn0 * CCH);
}

// ------------------------------------------------------------------ kernel 2
// WPB waves / block, one wave32 per point; uniform control flow across block.
__device__ __forceinline__ void ln_relu_rows(const float* src, _Float16* dst,
                                             const float* __restrict__ g,
                                             const float* __restrict__ bta) {
    int lane = threadIdx.x & 31;
    if (lane < 16) {
        const float* r = src + lane * 64;
        float s = 0.f;
        for (int c = 0; c < 64; ++c) s += r[c];
        float mean = s * (1.f / 64.f);
        float v = 0.f;
        for (int c = 0; c < 64; ++c) { float d = r[c] - mean; v += d * d; }
        float inv = rsqrtf(v * (1.f / 64.f) + EPS);
        _Float16* d = dst + lane * 64;
        for (int c = 0; c < 64; ++c) {
            float t = (r[c] - mean) * inv * g[c] + bta[c];
            d[c] = (_Float16)fmaxf(t, 0.f);
        }
    }
}

__device__ __forceinline__ void gemm_attn(const _Float16* ah, const _Float16* wf,
                                          const float* __restrict__ bias,
                                          float* outT) {
    int lane = threadIdx.x & 31, nn = lane & 15, hi = lane >> 4;
    v16h a0 = load_a_frag(ah, 64, 0);
    v16h a1 = load_a_frag(ah, 64, 32);
#pragma unroll
    for (int nt = 0; nt < 4; ++nt) {
        float bb = bias[nt * 16 + nn];
        v8f acc;
#pragma unroll
        for (int r = 0; r < 8; ++r) acc[r] = bb;
        acc = wmma_f16(a0, load_b_frag(wf + (nt * 2 + 0) * 512), acc);
        acc = wmma_f16(a1, load_b_frag(wf + (nt * 2 + 1) * 512), acc);
#pragma unroll
        for (int r = 0; r < 8; ++r)
            outT[(r + hi * 8) * 64 + nt * 16 + nn] = acc[r];
    }
}

__global__ __launch_bounds__(32 * WPB) void lfa_attn_kernel(
    const float* __restrict__ xyz, const float* __restrict__ qb,
    const _Float16* __restrict__ kb, const _Float16* __restrict__ vb,
    const int* __restrict__ nidx,
    const float* __restrict__ Wd1, const float* __restrict__ bd1,
    const float* __restrict__ lnd1_g, const float* __restrict__ lnd1_b,
    const float* __restrict__ Wd2, const float* __restrict__ bd2,
    const float* __restrict__ Wg1, const float* __restrict__ bg1,
    const float* __restrict__ lng1_g, const float* __restrict__ lng1_b,
    const float* __restrict__ Wg2, const float* __restrict__ bg2,
    const float* __restrict__ lng2_g, const float* __restrict__ lng2_b,
    const float* __restrict__ Wmlp, const float* __restrict__ bnm_g,
    const float* __restrict__ bnm_b, const float* __restrict__ bnm_m,
    const float* __restrict__ bnm_v, const float* __restrict__ feat,
    const float* __restrict__ Wsc, const float* __restrict__ bns_g,
    const float* __restrict__ bns_b, const float* __restrict__ bns_m,
    const float* __restrict__ bns_v, float* __restrict__ out) {
    __shared__ __align__(16) _Float16 wfrag[16 * 512];   // Wg1: 0..7, Wg2: 8..15
    __shared__ __align__(16) float    posT[WPB][16 * 64];
    __shared__ __align__(16) float    attnT[WPB][16 * 64];
    __shared__ __align__(16) _Float16 Ah[WPB][16 * 64];
    __shared__ __align__(16) _Float16 kstage[WPB][16 * 64]; // async k gather
    __shared__ __align__(16) _Float16 vstage[WPB][16 * 64]; // async v gather
    __shared__ float resv[WPB][64];
    __shared__ int   idxs[WPB][16];

    int tid = threadIdx.x;
    for (int idx = tid; idx < 16 * 512; idx += 32 * WPB) {
        int f = idx >> 9, r = idx & 511, lane = r >> 4, j = r & 15;
        int g = f & 7, nt = g >> 1, kc = g & 1;
        const float* W = (f < 8) ? Wg1 : Wg2;
        wfrag[idx] =
            (_Float16)W[(nt * 16 + (lane & 15)) * CCH + kc * 32 +
                        ((lane >> 4) ? 16 : 0) + j];
    }
    __syncthreads();

    int warp = tid >> 5, lane = tid & 31;
    int pt = blockIdx.x * WPB + warp;               // flat point id, always valid
    int b = pt / NPT, n = pt % NPT;
    float*    pos = posT[warp];
    float*    at  = attnT[warp];
    _Float16* ah  = Ah[warp];
    _Float16* kst = kstage[warp];
    _Float16* vst = vstage[warp];
    float*    res = resv[warp];
    int*      idv = idxs[warp];

    if (lane < 16) idv[lane] = nidx[((size_t)b * NPT + n) * KNN + lane];

    // ---- async gather of all 16 k rows + 16 v rows (128 B each) into LDS ----
    // Each wave issues 4+4 GLOBAL_LOAD_ASYNC_TO_LDS_B128 (32 lanes x 16 B =
    // 4 rows per issue). Overlapped with the pos-MLP below; fenced by
    // s_wait_asynccnt 0.
    {
        int sub = (lane & 7) * 16;                    // byte chunk within row
        int r0  = lane >> 3;                          // 0..3
        unsigned kl = lds_off(kst);
        unsigned vl = lds_off(vst);
#pragma unroll
        for (int i = 0; i < 4; ++i) {
            int m = i * 4 + r0;
            size_t rowoff = ((size_t)b * NPT + idv[m]) * CCH;
            async_copy_b128(kl + m * 128 + sub,
                            (unsigned long long)(kb + rowoff) + sub);
            async_copy_b128(vl + m * 128 + sub,
                            (unsigned long long)(vb + rowoff) + sub);
        }
    }

    // pos MLP: rel@Wd1^T -> LN(3) -> ReLU -> @Wd2^T + bd2  (lanes 0-15, row each)
    if (lane < 16) {
        int m = lane, j = idv[m];
        float cx = xyz[((size_t)b * NPT + n) * 3 + 0];
        float cy = xyz[((size_t)b * NPT + n) * 3 + 1];
        float cz = xyz[((size_t)b * NPT + n) * 3 + 2];
        float rx = cx - xyz[((size_t)b * NPT + j) * 3 + 0];
        float ry = cy - xyz[((size_t)b * NPT + j) * 3 + 1];
        float rz = cz - xyz[((size_t)b * NPT + j) * 3 + 2];
        float p0 = rx * Wd1[0] + ry * Wd1[1] + rz * Wd1[2] + bd1[0];
        float p1 = rx * Wd1[3] + ry * Wd1[4] + rz * Wd1[5] + bd1[1];
        float p2 = rx * Wd1[6] + ry * Wd1[7] + rz * Wd1[8] + bd1[2];
        float mean = (p0 + p1 + p2) * (1.f / 3.f);
        float d0 = p0 - mean, d1 = p1 - mean, d2 = p2 - mean;
        float inv = rsqrtf((d0 * d0 + d1 * d1 + d2 * d2) * (1.f / 3.f) + EPS);
        p0 = fmaxf(d0 * inv * lnd1_g[0] + lnd1_b[0], 0.f);
        p1 = fmaxf(d1 * inv * lnd1_g[1] + lnd1_b[1], 0.f);
        p2 = fmaxf(d2 * inv * lnd1_g[2] + lnd1_b[2], 0.f);
        for (int c = 0; c < 64; ++c)
            pos[m * 64 + c] =
                p0 * Wd2[c * 3 + 0] + p1 * Wd2[c * 3 + 1] + p2 * Wd2[c * 3 + 2] + bd2[c];
    }
    wait_asynccnt0();                                 // k/v staging complete
    __syncthreads();

    // a0 = q - kf + pos  (kf from LDS staging)
    {
        float q0 = qb[((size_t)b * NPT + n) * CCH + lane];
        float q1 = qb[((size_t)b * NPT + n) * CCH + lane + 32];
        for (int m = 0; m < KNN; ++m) {
            at[m * 64 + lane]      = q0 - (float)kst[m * 64 + lane]      + pos[m * 64 + lane];
            at[m * 64 + lane + 32] = q1 - (float)kst[m * 64 + lane + 32] + pos[m * 64 + lane + 32];
        }
    }
    __syncthreads();

    ln_relu_rows(at, ah, lng1_g, lng1_b);
    __syncthreads();
    gemm_attn(ah, wfrag, bg1, at);
    __syncthreads();
    ln_relu_rows(at, ah, lng2_g, lng2_b);
    __syncthreads();
    gemm_attn(ah, wfrag + 8 * 512, bg2, at);
    __syncthreads();

    // softmax over K per channel; weighted sum with (vf + pos), vf from LDS
#pragma unroll
    for (int t = 0; t < 2; ++t) {
        int c = lane + t * 32;
        float mx = -1e30f;
        for (int m = 0; m < KNN; ++m) mx = fmaxf(mx, at[m * 64 + c]);
        float s = 0.f, acc = 0.f;
        for (int m = 0; m < KNN; ++m) {
            float p  = __expf(at[m * 64 + c] - mx);
            float vv = (float)vst[m * 64 + c] + pos[m * 64 + c];
            s += p;
            acc += p * vv;
        }
        res[c] = acc / s;
    }
    __syncthreads();

    // epilogue: res@Wmlp^T -> BN-ReLU ; shortcut f@Wsc^T -> BN-ReLU ; leaky add
#pragma unroll
    for (int t = 0; t < 2; ++t) {
        int c = lane + t * 32;
        float acc = 0.f;
        for (int j2 = 0; j2 < 64; ++j2) acc += res[j2] * Wmlp[c * 64 + j2];
        float mm = fmaxf(
            bnm_g[c] * (acc - bnm_m[c]) * rsqrtf(bnm_v[c] + EPS) + bnm_b[c], 0.f);
        float sc = 0.f;
        for (int d = 0; d < DIN; ++d)
            sc += feat[((size_t)b * DIN + d) * NPT + n] * Wsc[c * DIN + d];
        sc = fmaxf(
            bns_g[c] * (sc - bns_m[c]) * rsqrtf(bns_v[c] + EPS) + bns_b[c], 0.f);
        float o = mm + sc;
        out[((size_t)b * CCH + c) * NPT + n] = (o < 0.f) ? 0.2f * o : o;
    }
}

// ------------------------------------------------------------------ launcher
extern "C" void kernel_launch(void* const* d_in, const int* in_sizes, int n_in,
                              void* d_out, int out_size, void* d_ws, size_t ws_size,
                              hipStream_t stream) {
    const float* feature = (const float*)d_in[0];
    const float* xyz     = (const float*)d_in[1];
    const float* W1  = (const float*)d_in[2];
    const float* b1  = (const float*)d_in[3];
    const float* Wq  = (const float*)d_in[4];
    const float* bq  = (const float*)d_in[5];
    const float* Wk  = (const float*)d_in[6];
    const float* bk  = (const float*)d_in[7];
    const float* Wv  = (const float*)d_in[8];
    const float* bv  = (const float*)d_in[9];
    const float* Wd1 = (const float*)d_in[10];
    const float* bd1 = (const float*)d_in[11];
    const float* lnd1_g = (const float*)d_in[12];
    const float* lnd1_b = (const float*)d_in[13];
    const float* Wd2 = (const float*)d_in[14];
    const float* bd2 = (const float*)d_in[15];
    const float* Wg1 = (const float*)d_in[16];
    const float* bg1 = (const float*)d_in[17];
    const float* lng1_g = (const float*)d_in[18];
    const float* lng1_b = (const float*)d_in[19];
    const float* Wg2 = (const float*)d_in[20];
    const float* bg2 = (const float*)d_in[21];
    const float* lng2_g = (const float*)d_in[22];
    const float* lng2_b = (const float*)d_in[23];
    const float* Wmlp  = (const float*)d_in[24];
    const float* bnm_g = (const float*)d_in[25];
    const float* bnm_b = (const float*)d_in[26];
    const float* bnm_m = (const float*)d_in[27];
    const float* bnm_v = (const float*)d_in[28];
    const float* Wsc   = (const float*)d_in[29];
    const float* bns_g = (const float*)d_in[30];
    const float* bns_b = (const float*)d_in[31];
    const float* bns_m = (const float*)d_in[32];
    const float* bns_v = (const float*)d_in[33];
    const int*   nidx  = (const int*)d_in[34];
    float* outp = (float*)d_out;

    const size_t NBN = (size_t)BQ * NPT;             // 40960 points
    float*    qb = (float*)d_ws;                     // f32: B*N*C
    _Float16* kb = (_Float16*)(qb + NBN * CCH);      // f16: B*N*C
    _Float16* vb = kb + NBN * CCH;                   // f16: B*N*C

    // kernel 1: 16 points per wave, 4 waves per block
    dim3 g1((unsigned)(NBN / 16 / 4)), blk1(128);
    lfa_qkv_kernel<<<g1, blk1, 0, stream>>>(feature, W1, b1, Wq, bq, Wk, bk, Wv, bv,
                                            qb, kb, vb);

    // kernel 2: one point per wave, WPB waves per block
    dim3 g2((unsigned)(NBN / WPB)), blk2(32 * WPB);
    lfa_attn_kernel<<<g2, blk2, 0, stream>>>(
        xyz, qb, kb, vb, nidx, Wd1, bd1, lnd1_g, lnd1_b, Wd2, bd2, Wg1, bg1,
        lng1_g, lng1_b, Wg2, bg2, lng2_g, lng2_b, Wmlp, bnm_g, bnm_b, bnm_m,
        bnm_v, feature, Wsc, bns_g, bns_b, bns_m, bns_v, outp);
}
